// HeteroGNN_21655225106538
// MI455X (gfx1250) — matrix operand order repaired and checked by reading, not compiled, not run on hardware
//
#include <hip/hip_runtime.h>
#include <hip/hip_bf16.h>

// ---------------------------------------------------------------------------
// Types
// ---------------------------------------------------------------------------
typedef __bf16 bf16;
typedef __attribute__((ext_vector_type(16))) __bf16 v16bf;
typedef __attribute__((ext_vector_type(8)))  __bf16 v8bf;
typedef __attribute__((ext_vector_type(4)))  __bf16 v4bf;
typedef __attribute__((ext_vector_type(8)))  float  v8f;
typedef __attribute__((ext_vector_type(4)))  float  f32x4;
typedef __attribute__((ext_vector_type(4)))  unsigned int u32x4;

#define GLDA 48    // LDS row stride (bf16) for A/B tiles  (96B, 16B aligned)
#define GLDC 132   // LDS row stride (f32) for C tile
#define TILE_BF (128 * GLDA)          // one A or B tile in bf16 elems
#define FLAG_RELU 1
#define FLAG_LN   2

// ---------------------------------------------------------------------------
// Fused WMMA GEMM:  out[M,128] = act( A[M,Kp](bf16) @ Wt^T + bias )
//   Kp MUST be a multiple of 32 (caller zero-pads K=16 operands to 32).
//   Wt is the weight pre-transposed/padded to [128][Kp] bf16.
//   flags: bit0 = ReLU, bit1 = LayerNorm over the 128 features (lng/lnb).
//   outF (f32) and outB (bf16) are each optional (nullptr to skip).
//   Double-buffered LDS staging. A sched_barrier between the batched B
//   fragment loads and the WMMA chain pins all 16 ds_loads before the 8
//   WMMAs: one s_wait_dscnt, then back-to-back WMMAs.
// ---------------------------------------------------------------------------
__global__ __launch_bounds__(256, 2)
void k_gemm128(const bf16* __restrict__ A, int M, int Kp,
               const bf16* __restrict__ Wt,
               const float* __restrict__ bias,
               const float* __restrict__ lng,
               const float* __restrict__ lnb,
               int flags,
               float* __restrict__ outF,
               bf16*  __restrict__ outB)
{
    extern __shared__ char smem[];
    bf16*  sT    = (bf16*)smem;                 // 4 tiles: [buf][A|B][128*GLDA]
    float* sC    = (float*)smem;                // aliased over sT (used after loop)
    float* sMean = sC + 128 * GLDC;
    float* sRstd = sMean + 128;

    const int tid  = threadIdx.x;
    const int lane = tid & 31;
    const int wv   = tid >> 5;      // wave id 0..7 -> rows wv*16..wv*16+15
    const int hi   = lane >> 4;     // half-wave select
    const int l16  = lane & 15;
    const int rowBase = blockIdx.x * 128;

    // staging role of this thread: one 16-bf16 (2x16B) chunk of A and of B
    const int r0  = tid >> 1;             // 0..127
    const int kk0 = (tid & 1) * 16;       // 0 or 16 (bf16 elems within 32-slab)
    int rowA = rowBase + r0;
    if (rowA >= M) rowA = M - 1;          // clamp: avoids exec-masked loads
    const long long aBase = (long long)rowA * Kp + kk0;
    const long long bBase = (long long)r0   * Kp + kk0;
    const int stOff = r0 * GLDA + kk0;

    v8f acc[8];
#pragma unroll
    for (int s = 0; s < 8; ++s)
#pragma unroll
        for (int j = 0; j < 8; ++j) acc[s][j] = 0.0f;

    const int nK = Kp >> 5;

    // ---- prologue: stage tile 0 into buffer 0 ----
    {
        u32x4 a0 = *(const u32x4*)(A  + aBase);
        u32x4 a1 = *(const u32x4*)(A  + aBase + 8);
        u32x4 b0 = *(const u32x4*)(Wt + bBase);
        u32x4 b1 = *(const u32x4*)(Wt + bBase + 8);
        bf16* dA = sT;
        bf16* dB = sT + TILE_BF;
        *(u32x4*)(dA + stOff)     = a0;
        *(u32x4*)(dA + stOff + 8) = a1;
        *(u32x4*)(dB + stOff)     = b0;
        *(u32x4*)(dB + stOff + 8) = b1;
    }
    __syncthreads();

    for (int t = 0; t < nK; ++t) {
        const int cur = t & 1;
        const bf16* sAc = sT + cur * (2 * TILE_BF);
        const bf16* sBc = sAc + TILE_BF;
        const bool pf = (t + 1) < nK;

        // ---- prefetch next K-slab from global while we compute ----
        u32x4 a0, a1, b0, b1;
        if (pf) {
            const long long ko = (long long)(t + 1) * 32;
            a0 = *(const u32x4*)(A  + aBase + ko);
            a1 = *(const u32x4*)(A  + aBase + ko + 8);
            b0 = *(const u32x4*)(Wt + bBase + ko);
            b1 = *(const u32x4*)(Wt + bBase + ko + 8);
        }

        // ---- A fragment (16x32 bf16) per documented lane layout ----
        const int am  = wv * 16 + l16;
        const int kc1 = hi * 8;        // lanes 0-15: K 0..7 ; 16-31: K 8..15
        const int kc2 = 16 + hi * 8;   // lanes 0-15: K16..23; 16-31: K24..31
        v8bf alo = *(const v8bf*)(sAc + am * GLDA + kc1);
        v8bf ahi = *(const v8bf*)(sAc + am * GLDA + kc2);
        v16bf afrag;
#pragma unroll
        for (int i = 0; i < 8; ++i) { afrag[i] = alo[i]; afrag[i + 8] = ahi[i]; }

        // ---- batch-load ALL 8 B fragments ----
        const int kc = hi * 8;                  // K base for this half-wave (lo: kc..kc+7, hi: kc+16..kc+23)
        v16bf bfr[8];
#pragma unroll
        for (int s = 0; s < 8; ++s) {
            const int bn = s * 16 + l16;        // N index
            v8bf blo = *(const v8bf*)(sBc + bn * GLDA + kc);
            v8bf bhi = *(const v8bf*)(sBc + bn * GLDA + kc + 16);
#pragma unroll
            for (int i = 0; i < 8; ++i) { bfr[s][i] = blo[i]; bfr[s][i + 8] = bhi[i]; }
        }

        // Hard scheduling fence: nothing may cross. All 16 ds_loads stay
        // above, all 8 WMMAs below -> single s_wait_dscnt, WMMAs back-to-back.
        __builtin_amdgcn_sched_barrier(0);

#pragma unroll
        for (int s = 0; s < 8; ++s)
            acc[s] = __builtin_amdgcn_wmma_f32_16x16x32_bf16(
                false, afrag, false, bfr[s], (short)0, acc[s], false, false);

        // ---- commit prefetched slab into the other buffer ----
        if (pf) {
            bf16* dA = sT + (cur ^ 1) * (2 * TILE_BF);
            bf16* dB = dA + TILE_BF;
            *(u32x4*)(dA + stOff)     = a0;
            *(u32x4*)(dA + stOff + 8) = a1;
            *(u32x4*)(dB + stOff)     = b0;
            *(u32x4*)(dB + stOff + 8) = b1;
        }
        __syncthreads();
    }

    // ---- epilogue: accumulators (+bias) -> LDS C tile (aliased) ----
#pragma unroll
    for (int s = 0; s < 8; ++s) {
        const int c = s * 16 + l16;
        const float bv = bias ? bias[c] : 0.0f;
#pragma unroll
        for (int j = 0; j < 8; ++j) {
            const int r = wv * 16 + hi * 8 + j;  // C layout: VGPR j -> M=j (+8 hi)
            sC[r * GLDC + c] = acc[s][j] + bv;
        }
    }
    __syncthreads();

    if (flags & FLAG_LN) {
        if (tid < 128) {
            const float* row = sC + tid * GLDC;
            float s1 = 0.0f, s2 = 0.0f;
#pragma unroll 8
            for (int c = 0; c < 128; ++c) { float v = row[c]; s1 += v; s2 += v * v; }
            const float mean = s1 * (1.0f / 128.0f);
            const float var  = s2 * (1.0f / 128.0f) - mean * mean;
            sMean[tid] = mean;
            sRstd[tid] = rsqrtf(var + 1e-5f);
        }
        __syncthreads();
    }

    {
        const int row  = tid >> 1;
        const int half = tid & 1;
        const int grow = rowBase + row;
        if (grow < M) {
            float mean = 0.0f, rstd = 1.0f;
            if (flags & FLAG_LN) { mean = sMean[row]; rstd = sRstd[row]; }
            const float* src = sC + row * GLDC;
            for (int c0 = 0; c0 < 64; c0 += 4) {
                const int c = half * 64 + c0;
                f32x4 v;
#pragma unroll
                for (int i = 0; i < 4; ++i) {
                    float x = src[c + i];
                    if (flags & FLAG_LN) x = (x - mean) * rstd * lng[c + i] + lnb[c + i];
                    if (flags & FLAG_RELU) x = fmaxf(x, 0.0f);
                    v[i] = x;
                }
                if (outF) *(f32x4*)(outF + (long long)grow * 128 + c) = v;
                if (outB) {
                    v4bf b;
#pragma unroll
                    for (int i = 0; i < 4; ++i) b[i] = (bf16)v[i];
                    *(v4bf*)(outB + (long long)grow * 128 + c) = b;
                }
            }
        }
    }
}

// ---------------------------------------------------------------------------
// Small helper kernels
// ---------------------------------------------------------------------------
// x [M][Kin] f32 -> y [M][Kp] bf16, zero-padded columns Kin..Kp
__global__ void k_f32_to_bf16_pad(const float* __restrict__ x, bf16* __restrict__ y,
                                  int M, int Kin, int Kp)
{
    long long total = (long long)M * Kp;
    for (long long i = blockIdx.x * 256ll + threadIdx.x; i < total;
         i += (long long)gridDim.x * 256) {
        int r = (int)(i / Kp), c = (int)(i % Kp);
        y[i] = (c < Kin) ? (bf16)x[(long long)r * Kin + c] : (bf16)0.0f;
    }
}

// W [K][128] f32 -> Wt [128][Kp] bf16 (transposed, zero-padded K..Kp)
__global__ void k_wt_convert(const float* __restrict__ W, bf16* __restrict__ Wt,
                             int K, int Kp)
{
    int idx = blockIdx.x * 256 + threadIdx.x;
    if (idx < 128 * Kp) {
        int n = idx / Kp, k = idx % Kp;
        Wt[idx] = (k < K) ? (bf16)W[k * 128 + n] : (bf16)0.0f;
    }
}

__global__ void k_fill_u32(unsigned* __restrict__ p, unsigned v, int n)
{
    int i = blockIdx.x * 256 + threadIdx.x;
    if (i < n) p[i] = v;
}

// h[e, 0:512] = bf16{ n[src[e]], n[dst[e]], e_feat[e], u[eb[e]] }
__global__ void k_build_hedge(const float* __restrict__ nF, const float* __restrict__ eF,
                              const float* __restrict__ uF,
                              const int* __restrict__ src, const int* __restrict__ dst,
                              const int* __restrict__ eb,
                              bf16* __restrict__ h, int E)
{
    int t = blockIdx.x * 256 + threadIdx.x;          // E*64 threads
    if (t >= E * 64) return;
    int e = t >> 6, g = t & 63;
    int seg = g >> 4, off = (g & 15) * 8;
    int row; const float* base;
    if (seg == 0)      { row = src[e]; base = nF; }
    else if (seg == 1) { row = dst[e]; base = nF; }
    else if (seg == 2) { row = e;      base = eF; }
    else               { row = eb[e];  base = uF; }
    const f32x4* p = (const f32x4*)(base + (long long)row * 128 + off);
    f32x4 a = p[0], b = p[1];
    v8bf o;
#pragma unroll
    for (int i = 0; i < 4; ++i) { o[i] = (bf16)a[i]; o[i + 4] = (bf16)b[i]; }
    *(v8bf*)(h + (long long)e * 512 + seg * 128 + off) = o;
}

// out[i, 0:384] = bf16{ a[i], b[i], c[idx ? idx[i] : i] }
__global__ void k_build_cat3(const float* __restrict__ A, const float* __restrict__ Bm,
                             const float* __restrict__ C, const int* __restrict__ idx,
                             bf16* __restrict__ out, int M)
{
    int t = blockIdx.x * 256 + threadIdx.x;          // M*48 threads
    if (t >= M * 48) return;
    int i = t / 48, g = t % 48;
    int seg = g >> 4, off = (g & 15) * 8;
    int row = i; const float* base;
    if (seg == 0)      base = A;
    else if (seg == 1) base = Bm;
    else               { base = C; row = idx ? idx[i] : i; }
    const f32x4* p = (const f32x4*)(base + (long long)row * 128 + off);
    f32x4 a = p[0], b = p[1];
    v8bf o;
#pragma unroll
    for (int k = 0; k < 4; ++k) { o[k] = (bf16)a[k]; o[k + 4] = (bf16)b[k]; }
    *(v8bf*)(out + (long long)i * 384 + seg * 128 + off) = o;
}

__device__ __forceinline__ unsigned enc_f32(float x)
{
    unsigned u = __float_as_uint(x);
    return (u & 0x80000000u) ? ~u : (u | 0x80000000u);
}
__device__ __forceinline__ float dec_f32(unsigned e)
{
    unsigned u = (e & 0x80000000u) ? (e ^ 0x80000000u) : ~e;
    return __uint_as_float(u);
}

// per-edge: sc = a . leaky_relu(gl[dst]+gr, 0.2); segment max into mseg (encoded)
__global__ __launch_bounds__(256)
void k_attn_score(const float* __restrict__ gl, const float* __restrict__ gr,
                  const float* __restrict__ avec, const int* __restrict__ edst,
                  float* __restrict__ sc, unsigned* __restrict__ mseg, int E)
{
    int e = blockIdx.x * 8 + (threadIdx.x >> 5);
    int lane = threadIdx.x & 31;
    if (e >= E) return;
    int d = edst[e];
    f32x4 a4 = *(const f32x4*)(avec + lane * 4);
    f32x4 g4 = *(const f32x4*)(gl + (long long)d * 128 + lane * 4);
    f32x4 r4 = *(const f32x4*)(gr + (long long)e * 128 + lane * 4);
    float s = 0.0f;
#pragma unroll
    for (int i = 0; i < 4; ++i) {
        float t = g4[i] + r4[i];
        t = t > 0.0f ? t : 0.2f * t;
        s += t * a4[i];
    }
#pragma unroll
    for (int off = 16; off > 0; off >>= 1) s += __shfl_xor(s, off, 32);
    if (lane == 0) {
        sc[e] = s;
        atomicMax(mseg + d, enc_f32(s));
    }
}

__global__ void k_attn_exp(const float* __restrict__ sc, const int* __restrict__ edst,
                           const unsigned* __restrict__ mseg,
                           float* __restrict__ ex, float* __restrict__ den, int E)
{
    int e = blockIdx.x * 256 + threadIdx.x;
    if (e >= E) return;
    int d = edst[e];
    float x = __expf(sc[e] - dec_f32(mseg[d]));
    ex[e] = x;
    atomicAdd(den + d, x);
}

__global__ __launch_bounds__(256)
void k_attn_agg(const float* __restrict__ gr, const float* __restrict__ ex,
                const float* __restrict__ den, const int* __restrict__ edst,
                float* __restrict__ agg, int E)
{
    int e = blockIdx.x * 8 + (threadIdx.x >> 5);
    int lane = threadIdx.x & 31;
    if (e >= E) return;
    int d = edst[e];
    float alpha = ex[e] / fmaxf(den[d], 1e-16f);
    f32x4 r4 = *(const f32x4*)(gr + (long long)e * 128 + lane * 4);
    float* dp = agg + (long long)d * 128 + lane * 4;
    atomicAdd(dp + 0, alpha * r4[0]);
    atomicAdd(dp + 1, alpha * r4[1]);
    atomicAdd(dp + 2, alpha * r4[2]);
    atomicAdd(dp + 3, alpha * r4[3]);
}

// dst[idx[r], :] += src[r, :]   (128-wide rows, wave per row)
__global__ __launch_bounds__(256)
void k_scatter_add(float* __restrict__ dst, const float* __restrict__ src,
                   const int* __restrict__ idx, int M)
{
    int r = blockIdx.x * 8 + (threadIdx.x >> 5);
    int lane = threadIdx.x & 31;
    if (r >= M) return;
    int d = idx[r];
    f32x4 s = *(const f32x4*)(src + (long long)r * 128 + lane * 4);
    float* dp = dst + (long long)d * 128 + lane * 4;
    atomicAdd(dp + 0, s[0]);
    atomicAdd(dp + 1, s[1]);
    atomicAdd(dp + 2, s[2]);
    atomicAdd(dp + 3, s[3]);
}

// ---------------------------------------------------------------------------
// Host orchestration
// ---------------------------------------------------------------------------
extern "C" void kernel_launch(void* const* d_in, const int* in_sizes, int n_in,
                              void* d_out, int out_size, void* d_ws, size_t ws_size,
                              hipStream_t stream)
{
    const int N = in_sizes[0] / 32;
    const int E = in_sizes[1] / 16;
    const int B = in_sizes[2] / 16;

    const float* node_x    = (const float*)d_in[0];
    const float* edge_attr = (const float*)d_in[1];
    const float* global_x  = (const float*)d_in[2];
    const int*   esrc      = (const int*)d_in[3];
    const int*   edst      = (const int*)d_in[4];
    const int*   nbatch    = (const int*)d_in[5];
    const int*   ebatch    = (const int*)d_in[6];
    auto F = [&](int i) { return (const float*)d_in[i]; };

    // params (jax pytree: dict keys sorted alphabetically)
    const float *attn_a = F(7), *attn_bo = F(8), *attn_wl = F(9), *attn_wo = F(10), *attn_wr = F(11);
    const float *ee_b1 = F(12), *ee_b2 = F(13), *ee_w1 = F(14), *ee_w2 = F(15);
    const float *eu_b1 = F(16), *eu_b2 = F(17), *eu_lnb = F(18), *eu_lng = F(19), *eu_w1 = F(20), *eu_w2 = F(21);
    const float *ge_b1 = F(22), *ge_b2 = F(23), *ge_w1 = F(24), *ge_w2 = F(25);
    const float *gu_b1 = F(26), *gu_b2 = F(27), *gu_b3 = F(28), *gu_b4 = F(29);
    const float *gu_ln1b = F(30), *gu_ln1g = F(31), *gu_ln2b = F(32), *gu_ln2g = F(33);
    const float *gu_w1 = F(34), *gu_w2 = F(35), *gu_w3 = F(36), *gu_w4 = F(37);
    const float *ne_b1 = F(38), *ne_b2 = F(39), *ne_w1 = F(40), *ne_w2 = F(41);

    // live outputs (f32) live directly in d_out: [n | e | u]
    float* nF = (float*)d_out;
    float* eF = nF + (size_t)N * 128;
    float* uF = eF + (size_t)E * 128;

    // ---- workspace bump allocator ----
    char* ws = (char*)d_ws;
    size_t off = 0;
    auto alloc = [&](size_t bytes) -> void* {
        void* p = ws + off;
        off += (bytes + 255) & ~(size_t)255;
        return p;
    };

    // transposed bf16 weights [128][Kp] (Kp = K padded to multiple of 32)
    bf16* wt_ne1 = (bf16*)alloc((size_t)32  * 128 * 2);
    bf16* wt_ne2 = (bf16*)alloc((size_t)128 * 128 * 2);
    bf16* wt_ee1 = (bf16*)alloc((size_t)32  * 128 * 2);   // K=16 padded to 32
    bf16* wt_ee2 = (bf16*)alloc((size_t)128 * 128 * 2);
    bf16* wt_ge1 = (bf16*)alloc((size_t)32  * 128 * 2);   // K=16 padded to 32
    bf16* wt_ge2 = (bf16*)alloc((size_t)128 * 128 * 2);
    bf16* wt_eu1 = (bf16*)alloc((size_t)512 * 128 * 2);
    bf16* wt_eu2 = (bf16*)alloc((size_t)128 * 128 * 2);
    bf16* wt_wl  = (bf16*)alloc((size_t)128 * 128 * 2);
    bf16* wt_wr  = (bf16*)alloc((size_t)128 * 128 * 2);
    bf16* wt_wo  = (bf16*)alloc((size_t)384 * 128 * 2);
    bf16* wt_gu1 = (bf16*)alloc((size_t)128 * 128 * 2);
    bf16* wt_gu2 = (bf16*)alloc((size_t)128 * 128 * 2);
    bf16* wt_gu3 = (bf16*)alloc((size_t)384 * 128 * 2);
    bf16* wt_gu4 = (bf16*)alloc((size_t)128 * 128 * 2);

    bf16* x_node_bf = (bf16*)alloc((size_t)N * 32 * 2);
    bf16* x_edge_bf = (bf16*)alloc((size_t)E * 32 * 2);   // padded 16 -> 32
    bf16* x_glob_bf = (bf16*)alloc((size_t)B * 32 * 2);   // padded 16 -> 32

    bf16* hid_bf  = (bf16*)alloc((size_t)E * 128 * 2);    // shared MLP-hidden buffer
    bf16* n_bf    = (bf16*)alloc((size_t)N * 128 * 2);
    bf16* e_bf    = (bf16*)alloc((size_t)E * 128 * 2);
    bf16* u_bf    = (bf16*)alloc((size_t)B * 128 * 2);
    bf16* h_edge  = (bf16*)alloc((size_t)E * 512 * 2);
    bf16* node_in = (bf16*)alloc((size_t)N * 384 * 2);
    bf16* glob_in = (bf16*)alloc((size_t)B * 384 * 2);

    float* glB  = (float*)alloc((size_t)N * 128 * 4);
    float* grB  = (float*)alloc((size_t)E * 128 * 4);
    float* aggB = (float*)alloc((size_t)N * 128 * 4);
    float* scB  = (float*)alloc((size_t)E * 4);
    float* exB  = (float*)alloc((size_t)E * 4);
    unsigned* msegB = (unsigned*)alloc((size_t)N * 4);
    float* denB = (float*)alloc((size_t)N * 4);
    float* u1B  = (float*)alloc((size_t)B * 128 * 4);
    float* naB  = (float*)alloc((size_t)B * 128 * 4);
    float* eaB  = (float*)alloc((size_t)B * 128 * 4);
    (void)ws_size; (void)n_in; (void)out_size;

    constexpr size_t TILE_BYTES = (size_t)4 * 128 * GLDA * 2;       // 49152
    constexpr size_t EPI_BYTES  = (size_t)128 * GLDC * 4 + 256 * 4; // 68608
    constexpr size_t GEMM_SMEM  = (TILE_BYTES > EPI_BYTES) ? TILE_BYTES : EPI_BYTES;

    auto gemm = [&](const bf16* A, int M, int Kp, const bf16* Wt, const float* bias,
                    const float* g, const float* b, int flags, float* oF, bf16* oB) {
        dim3 grid((M + 127) / 128);
        k_gemm128<<<grid, 256, GEMM_SMEM, stream>>>(A, M, Kp, Wt, bias, g, b, flags, oF, oB);
    };
    auto wtc = [&](const float* W, bf16* Wt, int K, int Kp) {
        int n = 128 * Kp;
        k_wt_convert<<<(n + 255) / 256, 256, 0, stream>>>(W, Wt, K, Kp);
    };
    auto f2bp = [&](const float* x, bf16* y, int M, int Kin, int Kp) {
        long long n = (long long)M * Kp;
        int blocks = (int)((n + 255) / 256); if (blocks > 8192) blocks = 8192;
        k_f32_to_bf16_pad<<<blocks, 256, 0, stream>>>(x, y, M, Kin, Kp);
    };

    // ---- one-time per launch: weight + input conversion ----
    wtc(ne_w1, wt_ne1, 32, 32);   wtc(ne_w2, wt_ne2, 128, 128);
    wtc(ee_w1, wt_ee1, 16, 32);   wtc(ee_w2, wt_ee2, 128, 128);
    wtc(ge_w1, wt_ge1, 16, 32);   wtc(ge_w2, wt_ge2, 128, 128);
    wtc(eu_w1, wt_eu1, 512, 512); wtc(eu_w2, wt_eu2, 128, 128);
    wtc(attn_wl, wt_wl, 128, 128); wtc(attn_wr, wt_wr, 128, 128);
    wtc(attn_wo, wt_wo, 384, 384);
    wtc(gu_w1, wt_gu1, 128, 128); wtc(gu_w2, wt_gu2, 128, 128);
    wtc(gu_w3, wt_gu3, 384, 384); wtc(gu_w4, wt_gu4, 128, 128);
    f2bp(node_x,    x_node_bf, N, 32, 32);
    f2bp(edge_attr, x_edge_bf, E, 16, 32);
    f2bp(global_x,  x_glob_bf, B, 16, 32);

    // ---- encoders ----
    gemm(x_node_bf, N, 32,  wt_ne1, ne_b1, nullptr, nullptr, FLAG_RELU, nullptr, hid_bf);
    gemm(hid_bf,    N, 128, wt_ne2, ne_b2, nullptr, nullptr, 0,         nF,      n_bf);
    gemm(x_edge_bf, E, 32,  wt_ee1, ee_b1, nullptr, nullptr, FLAG_RELU, nullptr, hid_bf);
    gemm(hid_bf,    E, 128, wt_ee2, ee_b2, nullptr, nullptr, 0,         eF,      e_bf);
    gemm(x_glob_bf, B, 32,  wt_ge1, ge_b1, nullptr, nullptr, FLAG_RELU, nullptr, hid_bf);
    gemm(hid_bf,    B, 128, wt_ge2, ge_b2, nullptr, nullptr, 0,         uF,      u_bf);

    // ---- message-passing rounds ----
    for (int round = 0; round < 3; ++round) {
        // edge update: h = [src,dst,e,u_e] -> MLP -> LN -> ReLU
        k_build_hedge<<<(E * 64 + 255) / 256, 256, 0, stream>>>(
            nF, eF, uF, esrc, edst, ebatch, h_edge, E);
        gemm(h_edge, E, 512, wt_eu1, eu_b1, nullptr, nullptr, FLAG_RELU, nullptr, hid_bf);
        gemm(hid_bf, E, 128, wt_eu2, eu_b2, eu_lng, eu_lnb, FLAG_LN | FLAG_RELU, eF, e_bf);

        // attention projections (n here is the pre-update node state)
        gemm(n_bf, N, 128, wt_wl, nullptr, nullptr, nullptr, 0, glB, nullptr);
        gemm(e_bf, E, 128, wt_wr, nullptr, nullptr, nullptr, 0, grB, nullptr);

        // segment softmax over edge_dst + weighted aggregation
        k_fill_u32<<<(N + 255) / 256, 256, 0, stream>>>(msegB, 0x007FFFFFu, N); // enc(-inf)
        hipMemsetAsync(denB, 0, (size_t)N * 4, stream);
        hipMemsetAsync(aggB, 0, (size_t)N * 128 * 4, stream);
        k_attn_score<<<(E + 7) / 8, 256, 0, stream>>>(glB, grB, attn_a, edst, scB, msegB, E);
        k_attn_exp<<<(E + 255) / 256, 256, 0, stream>>>(scB, edst, msegB, exB, denB, E);
        k_attn_agg<<<(E + 7) / 8, 256, 0, stream>>>(grB, exB, denB, edst, aggB, E);

        // node update: n = relu([n, agg, u_n] @ wo + bo)
        k_build_cat3<<<(N * 48 + 255) / 256, 256, 0, stream>>>(
            nF, aggB, uF, nbatch, node_in, N);
        gemm(node_in, N, 384, wt_wo, attn_bo, nullptr, nullptr, FLAG_RELU, nF, n_bf);

        // global update
        gemm(u_bf,   B, 128, wt_gu1, gu_b1, nullptr, nullptr, FLAG_RELU, nullptr, hid_bf);
        gemm(hid_bf, B, 128, wt_gu2, gu_b2, gu_ln1g, gu_ln1b, FLAG_LN, u1B, nullptr);
        hipMemsetAsync(naB, 0, (size_t)B * 128 * 4, stream);
        hipMemsetAsync(eaB, 0, (size_t)B * 128 * 4, stream);
        k_scatter_add<<<(N + 7) / 8, 256, 0, stream>>>(naB, nF, nbatch, N);
        k_scatter_add<<<(E + 7) / 8, 256, 0, stream>>>(eaB, eF, ebatch, E);
        k_build_cat3<<<(B * 48 + 255) / 256, 256, 0, stream>>>(
            u1B, naB, eaB, nullptr, glob_in, B);
        gemm(glob_in, B, 384, wt_gu3, gu_b3, nullptr, nullptr, FLAG_RELU, nullptr, hid_bf);
        gemm(hid_bf,  B, 128, wt_gu4, gu_b4, gu_ln2g, gu_ln2b, FLAG_LN | FLAG_RELU, uF, u_bf);
    }
}
